// StackLSTM_58282706207124
// MI455X (gfx1250) — matrix operand order repaired and checked by reading, not compiled, not run on hardware
//
#include <hip/hip_runtime.h>
#include <hip/hip_bf16.h>
#include <math.h>

typedef __bf16 bf16_t;
typedef __attribute__((ext_vector_type(8)))  __bf16 bf16x8;
typedef __attribute__((ext_vector_type(16))) __bf16 bf16x16;
typedef __attribute__((ext_vector_type(8)))  float  f32x8;

#define TDIM 1024
#define IDIM 1024
#define HDIM 1024
#define SDIM 64
#define NBLK 96
#define LROW 1032   // bf16 elems per LDS weight row (1024 + 8 pad -> 2064B stride)

// ---------------- helpers ----------------

__device__ inline float sigmoidf_(float x) { return 1.0f / (1.0f + __expf(-x)); }

// 16-bit A frag from a [.,K]-contiguous row: elems 0..15 = K+16h..K+16h+15
// (contiguous k order, matching the B^T fragments).
__device__ inline bf16x16 load_frag_rowmajor(const bf16_t* __restrict__ p, int half) {
  bf16x8 lo = *(const bf16x8*)(p + 16 * half);
  bf16x8 hi = *(const bf16x8*)(p + 16 * half + 8);
  bf16x16 v;
#pragma unroll
  for (int i = 0; i < 8; ++i) { v[i] = lo[i]; v[i + 8] = hi[i]; }
  return v;
}

__device__ inline bf16x16 load_bfrag_T(const bf16_t* __restrict__ rowptr, int half) {
  bf16x8 lo = *(const bf16x8*)(rowptr + 16 * half);
  bf16x8 hi = *(const bf16x8*)(rowptr + 16 * half + 8);
  bf16x16 v;
#pragma unroll
  for (int i = 0; i < 8; ++i) { v[i] = lo[i]; v[i + 8] = hi[i]; }
  return v;
}

__device__ inline void grid_barrier(unsigned* bar, unsigned nblk) {
  __syncthreads();
  if (threadIdx.x == 0) {
    unsigned* cnt = bar;
    unsigned* gen = bar + 1;
    __threadfence();
    unsigned g = atomicAdd(gen, 0u);
    if (atomicAdd(cnt, 1u) == nblk - 1u) {
      atomicExch(cnt, 0u);
      atomicAdd(gen, 1u);
    } else {
      while (atomicAdd(gen, 0u) == g) { __builtin_amdgcn_s_sleep(1); }
    }
    __threadfence();
  }
  __syncthreads();
}

// ---------------- prep kernels ----------------

__global__ void k_cast(const float* __restrict__ src, bf16_t* __restrict__ dst, int n) {
  int i = blockIdx.x * blockDim.x + threadIdx.x;
  if (i < n) dst[i] = (bf16_t)src[i];
}

__global__ void k_transpose10(const float* s0, const float* s1, const float* s2,
                              const float* s3, const float* s4, const float* s5,
                              const float* s6, const float* s7, const float* s8,
                              const float* s9, bf16_t* __restrict__ dst) {
  const float* srcs[10] = {s0, s1, s2, s3, s4, s5, s6, s7, s8, s9};
  const float* src = srcs[blockIdx.z];
  bf16_t* d = dst + (size_t)blockIdx.z * HDIM * HDIM;
  __shared__ float tile[32][33];
  int x0 = blockIdx.x * 32, y0 = blockIdx.y * 32;
  for (int j = threadIdx.y; j < 32; j += blockDim.y)
    tile[j][threadIdx.x] = src[(size_t)(y0 + j) * HDIM + x0 + threadIdx.x];
  __syncthreads();
  for (int j = threadIdx.y; j < 32; j += blockDim.y)
    d[(size_t)(x0 + j) * HDIM + y0 + threadIdx.x] = (bf16_t)tile[threadIdx.x][j];
}

__global__ void k_init(const float* __restrict__ h0, const float* __restrict__ c0,
                       const int* __restrict__ ops, float* __restrict__ hstack,
                       float* __restrict__ cstack, bf16_t* __restrict__ hstack_bf,
                       bf16_t* __restrict__ cstack_bf, int* __restrict__ pts,
                       float* __restrict__ biop, unsigned* __restrict__ bar) {
  int i = blockIdx.x * blockDim.x + threadIdx.x;
  const int total = (SDIM + 1) * HDIM;
  if (i < total) {
    float hv = (i < HDIM) ? h0[i] : 0.0f;
    float cv = (i < HDIM) ? c0[i] : 0.0f;
    hstack[i] = hv;  cstack[i] = cv;
    hstack_bf[i] = (bf16_t)hv;  cstack_bf[i] = (bf16_t)cv;
  }
  if (i == 0) {
    bar[0] = 0u; bar[1] = 0u;
    int acc = 0;
    for (int t = 0; t < TDIM; ++t) {
      acc += ops[t];
      pts[t] = acc;                          // pts = cumsum(ops)[:T]
      int o = ops[t + 1];
      biop[t] = (float)(o < 0 ? 0 : (o > 1 ? 1 : o));
    }
  }
}

// ---------------- phase 1: input projections via WMMA ----------------
__global__ void __launch_bounds__(128)
k_xproj(const bf16_t* __restrict__ Xb, const bf16_t* __restrict__ WxT,
        const float* __restrict__ bi, const float* __restrict__ bff,
        const float* __restrict__ bc, const float* __restrict__ bo,
        float* __restrict__ xg) {
  const int wave = threadIdx.x >> 5, lane = threadIdx.x & 31;
  const int col = lane & 15, half = lane >> 4;
  const int z = blockIdx.z;
  const int m0 = blockIdx.y * 16;
  const int n0 = (blockIdx.x * 4 + wave) * 64;
  const bf16_t* WT = WxT + (size_t)z * IDIM * HDIM;
  const float* bias = (z == 0) ? bi : (z == 1) ? bff : (z == 2) ? bc : bo;
  float* outp = xg + (size_t)z * TDIM * HDIM;

  f32x8 acc[4] = {};
  const bf16_t* arow = Xb + (size_t)(m0 + col) * IDIM;
  for (int k0 = 0; k0 < IDIM; k0 += 32) {
    bf16x16 a = load_frag_rowmajor(arow + k0, half);
#pragma unroll
    for (int j = 0; j < 4; ++j) {
      const bf16_t* brow = WT + (size_t)(n0 + 16 * j + col) * IDIM + k0;
      bf16x16 b = load_bfrag_T(brow, half);
      acc[j] = __builtin_amdgcn_wmma_f32_16x16x32_bf16(
          false, a, false, b, (short)0, acc[j], false, false);
    }
  }
#pragma unroll
  for (int j = 0; j < 4; ++j) {
    int n = n0 + 16 * j + col;
    float bv = bias[n];
#pragma unroll
    for (int r = 0; r < 8; ++r) {
      int m = m0 + r + 8 * half;
      outp[(size_t)m * HDIM + n] = acc[j][r] + bv;
    }
  }
}

// ---------------- phase 2: persistent recurrent kernel ----------------
// 96 blocks x 128 thr = 384 waves. Each wave owns a 16-out x 1024-K bf16 weight
// strip, staged ONCE into LDS via global_load_async_to_lds_b128 (ASYNCcnt),
// then 1024 steps of WMMA matvec read it via ds_load_b128 (conflict-free pad).
extern __shared__ bf16_t smem[];

__global__ void __launch_bounds__(128)
k_recurrent(const bf16_t* __restrict__ WaT, const bf16_t* __restrict__ WbT,
            const float* __restrict__ xg, float* __restrict__ hstack,
            float* __restrict__ cstack, bf16_t* __restrict__ hstack_bf,
            bf16_t* __restrict__ cstack_bf, const int* __restrict__ pts,
            const float* __restrict__ biop, float* __restrict__ avec,
            float* __restrict__ bvec, unsigned* __restrict__ bar,
            float* __restrict__ out) {
  const int lane = threadIdx.x & 31;
  const int wave = threadIdx.x >> 5;
  const int col = lane & 15, half = lane >> 4;
  const int gtid = blockIdx.x * blockDim.x + threadIdx.x;
  const int gwave = gtid >> 5;              // 0..383
  const int side = (gwave >= 192) ? 1 : 0;  // 0: h-side, 1: c-side
  const int n0 = (side ? gwave - 192 : gwave) * 16;
  const bf16_t* WT = side ? WbT : WaT;
  float* gvec = side ? bvec : avec;
  const float* xi = xg;
  const float* xf = xg + 1ull * TDIM * HDIM;
  const float* xc = xg + 2ull * TDIM * HDIM;
  const float* xo = xg + 3ull * TDIM * HDIM;

  // ---- one-time async stage of this wave's 16x1024 bf16 weight strip to LDS ----
  const unsigned lds_wave_b = (unsigned)(wave * 16 * LROW * 2);  // byte base
  for (int idx = lane; idx < 16 * 128; idx += 32) {              // 16 rows x 128 16B-chunks
    int r = idx >> 7, kc = idx & 127;
    const bf16_t* gsrc = WT + (size_t)(n0 + r) * HDIM + kc * 8;
    unsigned ldst = lds_wave_b + (unsigned)(r * LROW * 2 + kc * 16);
    asm volatile("global_load_async_to_lds_b128 %0, %1, off"
                 :: "v"(ldst), "v"((unsigned long long)(uintptr_t)gsrc)
                 : "memory");
  }
#if __has_builtin(__builtin_amdgcn_s_wait_asynccnt)
  __builtin_amdgcn_s_wait_asynccnt(0);
#else
  asm volatile("s_wait_asynccnt 0x0" ::: "memory");
#endif
  __syncthreads();

  const bf16_t* wrow_l = smem + (size_t)wave * 16 * LROW + (size_t)col * LROW;

  bf16x16 zero_a;
#pragma unroll
  for (int i = 0; i < 16; ++i) zero_a[i] = (bf16_t)0.0f;

  for (int t = 0; t < TDIM; ++t) {
    const int pt = pts[t];
    const bf16_t* vecb = (side ? cstack_bf : hstack_bf) + (size_t)pt * HDIM;
    f32x8 acc = {};
    for (int k0 = 0; k0 < HDIM; k0 += 32) {
      bf16x16 a = zero_a;
      if (col == 0)                        // only lanes 0/16 carry A row 0
        a = load_frag_rowmajor(vecb + k0, half);
      // B frag from LDS: 32B contiguous at row 'col', offset k0 + 16*half
      const bf16_t* lp = wrow_l + k0 + 16 * half;
      bf16x8 lo = *(const bf16x8*)lp;
      bf16x8 hi = *(const bf16x8*)(lp + 8);
      bf16x16 b;
#pragma unroll
      for (int i = 0; i < 8; ++i) { b[i] = lo[i]; b[i + 8] = hi[i]; }
      acc = __builtin_amdgcn_wmma_f32_16x16x32_bf16(
          false, a, false, b, (short)0, acc, false, false);
    }
    if (lane < 16) gvec[n0 + lane] = acc[0];   // D row 0 = VGPR0, lanes 0..15
    grid_barrier(bar, NBLK);

    if (gtid < HDIM) {
      const int n = gtid;
      const size_t tx = (size_t)t * HDIM + n;
      float a0 = avec[n], a1 = avec[HDIM + n], a2 = avec[2 * HDIM + n];
      float b0 = bvec[n], b1 = bvec[HDIM + n], b2 = bvec[2 * HDIM + n];
      float ig = sigmoidf_(xi[tx] + a0 + b0);
      float fg = sigmoidf_(xf[tx] + a1 + b1);
      float cv = cstack[(size_t)pt * HDIM + n];
      float cand = tanhf(xc[tx] + a1);      // source bug: candidate uses h@Whf
      float cg = fg * cv + ig * cand;
      float og = sigmoidf_(xo[tx] + a2 + b2);
      float nh = og * tanhf(cg);
      const size_t wr = (size_t)(pt + 1) * HDIM + n;
      hstack[wr] = nh;        cstack[wr] = cg;
      hstack_bf[wr] = (bf16_t)nh;  cstack_bf[wr] = (bf16_t)cg;
      float prev = hstack[(size_t)(pt - 1) * HDIM + n];
      float op = biop[t];
      out[tx] = op * nh + (1.0f - op) * prev;
    }
    grid_barrier(bar, NBLK);
  }
}

// ---------------- launcher ----------------

extern "C" void kernel_launch(void* const* d_in, const int* in_sizes, int n_in,
                              void* d_out, int out_size, void* d_ws, size_t ws_size,
                              hipStream_t stream) {
  (void)in_sizes; (void)n_in; (void)out_size; (void)ws_size;
  const float* inputs = (const float*)d_in[0];
  const float* h0  = (const float*)d_in[1];
  const float* c0  = (const float*)d_in[2];
  const float* Wxi = (const float*)d_in[3];
  const float* bxi = (const float*)d_in[4];
  const float* Whi = (const float*)d_in[5];
  const float* Wci = (const float*)d_in[6];
  const float* Wxf = (const float*)d_in[7];
  const float* bxf = (const float*)d_in[8];
  const float* Whf = (const float*)d_in[9];
  const float* Wcf = (const float*)d_in[10];
  const float* Wxc = (const float*)d_in[11];
  const float* bxc = (const float*)d_in[12];
  /* d_in[13] = Whc : unused (faithful to source bug) */
  const float* Wxo = (const float*)d_in[14];
  const float* bxo = (const float*)d_in[15];
  const float* Who = (const float*)d_in[16];
  const float* Wco = (const float*)d_in[17];
  const int*   ops = (const int*)d_in[18];
  float* out = (float*)d_out;

  char* ws = (char*)d_ws;
  size_t off = 0;
  auto alloc = [&](size_t bytes) -> char* {
    char* p = ws + off;
    off += (bytes + 255) & ~(size_t)255;
    return p;
  };
  bf16_t* Xb     = (bf16_t*)alloc((size_t)TDIM * IDIM * 2);          //  2 MB
  bf16_t* WT     = (bf16_t*)alloc(10ull * 1024 * 1024 * 2);          // 20 MB
  float*  xg     = (float*)alloc(4ull * TDIM * HDIM * 4);            // 16 MB
  float*  hstack = (float*)alloc((size_t)(SDIM + 1) * HDIM * 4);
  float*  cstack = (float*)alloc((size_t)(SDIM + 1) * HDIM * 4);
  bf16_t* hstack_bf = (bf16_t*)alloc((size_t)(SDIM + 1) * HDIM * 2);
  bf16_t* cstack_bf = (bf16_t*)alloc((size_t)(SDIM + 1) * HDIM * 2);
  float*  avec   = (float*)alloc(3 * HDIM * 4);
  float*  bvec   = (float*)alloc(3 * HDIM * 4);
  int*    pts    = (int*)alloc(TDIM * 4);
  float*  biop   = (float*)alloc(TDIM * 4);
  unsigned* bar  = (unsigned*)alloc(256);

  k_cast<<<(TDIM * IDIM + 255) / 256, 256, 0, stream>>>(inputs, Xb, TDIM * IDIM);
  k_transpose10<<<dim3(32, 32, 10), dim3(32, 8), 0, stream>>>(
      Wxi, Wxf, Wxc, Wxo, Whi, Whf, Who, Wci, Wcf, Wco, WT);
  k_init<<<((SDIM + 1) * HDIM + 255) / 256, 256, 0, stream>>>(
      h0, c0, ops, hstack, cstack, hstack_bf, cstack_bf, pts, biop, bar);

  const bf16_t* WxT = WT;                       // Wxi^T,Wxf^T,Wxc^T,Wxo^T
  const bf16_t* WaT = WT + 4ull * 1024 * 1024;  // Whi^T,Whf^T,Who^T
  const bf16_t* WbT = WT + 7ull * 1024 * 1024;  // Wci^T,Wcf^T,Wco^T

  k_xproj<<<dim3(4, 64, 4), 128, 0, stream>>>(Xb, WxT, bxi, bxf, bxc, bxo, xg);

  const size_t lds_bytes = 4ull * 16 * LROW * 2;   // 132096 B (<320KB/WGP)
  k_recurrent<<<NBLK, 128, lds_bytes, stream>>>(
      WaT, WbT, xg, hstack, cstack, hstack_bf, cstack_bf, pts, biop,
      avec, bvec, bar, out);
}
